// AdaMemorySnowFlakeNet_TokenBasedClassTokenizer_72756745994679
// MI455X (gfx1250) — matrix operand order repaired and checked by reading, not compiled, not run on hardware
//
#include <hip/hip_runtime.h>
#include <hip/hip_bf16.h>

// ---------------- problem constants (match reference) ----------------
constexpr int Bc  = 8;     // batch
constexpr int Cc  = 512;   // channels
constexpr int Nn  = 2048;  // points
constexpr int Dd  = 64;    // attn dim
constexpr int Kn  = 16;    // knn
constexpr int PHc = 64;    // pos hidden
constexpr int AHc = 256;   // attn hidden (4*D)

typedef __attribute__((ext_vector_type(16))) __bf16 v16bf;
typedef __attribute__((ext_vector_type(8)))  float  v8f;
typedef __attribute__((ext_vector_type(2)))  float  v2f;

// gfx1250 async global->LDS copy (ASYNCcnt path), guarded so the build stays
// green on toolchains that lack the builtin.
#if defined(__has_builtin)
#if __has_builtin(__builtin_amdgcn_global_load_async_to_lds_b128)
#define USE_ASYNC_LDS 1
#endif
#endif

#ifdef USE_ASYNC_LDS
typedef int v4i __attribute__((vector_size(16)));
typedef __attribute__((address_space(1))) v4i* gv4i_p;   // global int4*
typedef __attribute__((address_space(3))) v4i* lv4i_p;   // LDS int4*
__device__ __forceinline__ void async_copy16(const void* g, void* l) {
  __builtin_amdgcn_global_load_async_to_lds_b128(
      (gv4i_p)(unsigned long long)g,
      (lv4i_p)(unsigned int)(unsigned long long)l, 0, 0);
}
__device__ __forceinline__ void async_wait0() {
#if __has_builtin(__builtin_amdgcn_s_wait_asynccnt)
  __builtin_amdgcn_s_wait_asynccnt(0);
#else
  asm volatile("s_wait_asynccnt 0x0" ::: "memory");
#endif
}
#endif

// ---------------- small casts ----------------
__global__ void castf2bf_kernel(const float* __restrict__ src,
                                __bf16* __restrict__ dst, long n) {
  long i = (long)blockIdx.x * blockDim.x + threadIdx.x;
  if (i < n) dst[i] = (__bf16)src[i];
}

// kq = concat([key, query], channel) -> bf16 (B, 2C, N)
__global__ void cast_concat_kernel(const float* __restrict__ key,
                                   const float* __restrict__ query,
                                   __bf16* __restrict__ kq) {
  long i = (long)blockIdx.x * blockDim.x + threadIdx.x;
  long total = (long)Bc * 2 * Cc * Nn;
  if (i >= total) return;
  long n = i % Nn;
  long c = (i / Nn) % (2 * Cc);
  long b = i / ((long)Nn * 2 * Cc);
  float v = (c < Cc) ? key[((long)b * Cc + c) * Nn + n]
                     : query[((long)b * Cc + (c - Cc)) * Nn + n];
  kq[i] = (__bf16)v;
}

// ---------------- generic bf16 WMMA GEMM ----------------
// out[b,m,n] = epilogue( sum_k W[m,k] * X[b,k,n] + bias[m] (+ res) )
// Block tile 64x64, 4 waves (128 thr), each wave a 32x32 quadrant via
// 2x2 grid of v_wmma_f32_16x16x32_bf16 accumulators. K step 32.
// A tile staged with GLOBAL_LOAD_ASYNC_TO_LDS (contiguous 16B chunks);
// B tile staged synchronously because it is transposed into LDS.
__global__ __launch_bounds__(128)
void gemm_bf16_wmma(const __bf16* __restrict__ W,    // (M, Kd) row-major
                    const __bf16* __restrict__ X,    // (b: xbs) x (Kd, N)
                    const float* __restrict__ bias,  // (M)
                    const float* __restrict__ res,   // optional (b: obs)x(M,N)
                    float*  __restrict__ outF,       // optional
                    __bf16* __restrict__ outB,       // optional
                    int M, int Kd, int N, long xbs, long obs, int relu) {
  const int b   = blockIdx.z;
  const __bf16* Xb = X + (long)b * xbs;
  const int tm  = blockIdx.y * 64;
  const int tn  = blockIdx.x * 64;
  const int tid = threadIdx.x;
  const int wave = tid >> 5, lane = tid & 31;
  const int hf = lane >> 4, l15 = lane & 15;
  const int wm = (wave >> 1) * 32, wn = (wave & 1) * 32;

  __shared__ __bf16 lA[64][32];   // [m][k]
  __shared__ __bf16 lB[64][32];   // transposed: [n][k]

  v8f acc[2][2] = {};

  for (int k0 = 0; k0 < Kd; k0 += 32) {
#ifdef USE_ASYNC_LDS
    // A tile: 64 rows x 32 bf16 = 256 chunks of 16B, DMA'd straight to LDS.
    for (int t = tid; t < 256; t += 128) {
      int m = t >> 2, c = t & 3;
      async_copy16(&W[(long)(tm + m) * Kd + (k0 + c * 8)], &lA[m][c * 8]);
    }
#else
    for (int t = tid; t < 64 * 32; t += 128) {
      int m = t >> 5, k = t & 31;
      lA[m][k] = W[(long)(tm + m) * Kd + (k0 + k)];
    }
#endif
    for (int t = tid; t < 64 * 32; t += 128) {
      int n = t >> 5, k = t & 31;
      lB[n][k] = Xb[(long)(k0 + k) * N + (tn + n)];
    }
    if (k0 + 32 < Kd) {   // hint next tiles toward L2 (global_prefetch_b8)
      __builtin_prefetch(&W[(long)(tm + (tid >> 1)) * Kd + k0 + 32], 0, 0);
      __builtin_prefetch(&Xb[(long)(k0 + 32 + (tid >> 2)) * N + tn], 0, 0);
    }
#ifdef USE_ASYNC_LDS
    async_wait0();
#endif
    __syncthreads();
#pragma unroll
    for (int i = 0; i < 2; ++i) {
      v16bf afrag;
      const int m0 = wm + i * 16 + l15;
#pragma unroll
      for (int e = 0; e < 16; ++e) {   // A 16x32: lanes0-15 K=0..15, 16-31 K=16..31
        int kk = hf * 16 + e;
        afrag[e] = lA[m0][kk];
      }
#pragma unroll
      for (int j = 0; j < 2; ++j) {
        v16bf bfrag;
        const int n0 = wn + j * 16 + l15;
#pragma unroll
        for (int e = 0; e < 16; ++e)   // B 32x16: lanes0-15 K=0..15, 16-31 K=16..31
          bfrag[e] = lB[n0][hf * 16 + e];
        acc[i][j] = __builtin_amdgcn_wmma_f32_16x16x32_bf16(
            false, afrag, false, bfrag, (short)0, acc[i][j], false, false);
      }
    }
    __syncthreads();
  }
#pragma unroll
  for (int i = 0; i < 2; ++i)
#pragma unroll
    for (int j = 0; j < 2; ++j)
#pragma unroll
      for (int r = 0; r < 8; ++r) {
        int m = tm + wm + i * 16 + hf * 8 + r;   // C/D layout: row = half*8+r
        int n = tn + wn + j * 16 + l15;
        float v = acc[i][j][r] + bias[m];
        long o = (long)b * obs + (long)m * N + n;
        if (res)  v += res[o];
        if (relu) v = v > 0.f ? v : 0.f;
        if (outF) outF[o] = v;
        if (outB) outB[o] = (__bf16)v;
      }
}

// ---------------- KNN: thread per point, sorted top-16 insert ----------------
__global__ __launch_bounds__(256)
void knn_kernel(const float* __restrict__ pos, int* __restrict__ idx) {
  int n = blockIdx.x * blockDim.x + threadIdx.x;
  int b = blockIdx.y;
  if (n >= Nn) return;
  const float* pb = pos + (long)b * 3 * Nn;
  float px = pb[n], py = pb[Nn + n], pz = pb[2 * Nn + n];
  float bd[Kn]; int bi[Kn];
#pragma unroll
  for (int i = 0; i < Kn; ++i) { bd[i] = 3.4e38f; bi[i] = 0; }
  for (int m = 0; m < Nn; ++m) {
    float dx = px - pb[m], dy = py - pb[Nn + m], dz = pz - pb[2 * Nn + m];
    float d = dx * dx + dy * dy + dz * dz;
    if (d < bd[Kn - 1]) {
      float cd = d; int ci = m;
#pragma unroll
      for (int j = 0; j < Kn; ++j) {
        if (cd < bd[j]) {
          float td = bd[j]; int ti = bi[j];
          bd[j] = cd; bi[j] = ci; cd = td; ci = ti;
        }
      }
    }
  }
#pragma unroll
  for (int i = 0; i < Kn; ++i) idx[((long)b * Nn + n) * Kn + i] = bi[i];
}

// ---------------- fused per-point attention (one wave per point) ----------------
// pos_mlp via f32 WMMA (K=3 pad 4), wp2/wa1/wa2 via bf16 WMMA, softmax over
// the 16 neighbor columns with intra-half-wave shuffles, agg -> bf16.
__global__ __launch_bounds__(128)
void attn_point_kernel(const float* __restrict__ pos,
                       const int*   __restrict__ idx,
                       const float* __restrict__ qmat,   // (B,D,N) f32
                       const float* __restrict__ kmat,   // (B,D,N) f32
                       const float* __restrict__ vmat,   // (B,D,N) f32
                       const float* __restrict__ wp1, const float* __restrict__ bp1,
                       const __bf16* __restrict__ wp2bf, const float* __restrict__ bp2,
                       const __bf16* __restrict__ wa1bf, const float* __restrict__ ba1,
                       const __bf16* __restrict__ wa2bf, const float* __restrict__ ba2,
                       __bf16* __restrict__ aggbf) {    // (B,D,N) bf16
  const int tid = threadIdx.x;
  const int p   = tid >> 5;              // wave -> point slot
  const int lane = tid & 31;
  const int hf = lane >> 4, l15 = lane & 15;
  const int n = blockIdx.x * 4 + p;
  const int b = blockIdx.y;

  __shared__ int    s_idx[4][Kn];
  __shared__ float  s_posrel[4][4][Kn];        // K rows 0..3 (row 3 zero pad)
  __shared__ float  s_pe[4][Dd][Kn];           // pos embedding f32
  __shared__ __bf16 s_bop[4][Dd][Kn];          // B operand staging (pe_h, qk+pe)
  __shared__ __bf16 s_hid[4][AHc][Kn];         // attn hidden

  const float* pb = pos + (long)b * 3 * Nn;

  // step 1: gather neighbors (lanes 0..15)
  if (hf == 0) {
    int j  = l15;
    int nb = idx[((long)b * Nn + n) * Kn + j];
    s_idx[p][j] = nb;
    s_posrel[p][0][j] = pb[n] - pb[nb];
    s_posrel[p][1][j] = pb[Nn + n] - pb[Nn + nb];
    s_posrel[p][2][j] = pb[2 * Nn + n] - pb[2 * Nn + nb];
    s_posrel[p][3][j] = 0.f;
  }
  __syncthreads();

  // step 2: pe_h = relu(wp1 @ pos_rel + bp1) via v_wmma_f32_16x16x4_f32
  v2f bpos;
  bpos[0] = s_posrel[p][hf * 2 + 0][l15];      // f32 B 4x16: K = half*2 + e
  bpos[1] = s_posrel[p][hf * 2 + 1][l15];
#pragma unroll
  for (int t = 0; t < 4; ++t) {                // PH=64 -> 4 M tiles
    v2f af;
    int m  = 16 * t + l15;
    int k0 = hf * 2 + 0, k1 = hf * 2 + 1;      // f32 A 16x4: K = half*2 + e
    af[0] = (k0 < 3) ? wp1[m * 3 + k0] : 0.f;
    af[1] = (k1 < 3) ? wp1[m * 3 + k1] : 0.f;
    v8f acc = {};
    acc = __builtin_amdgcn_wmma_f32_16x16x4_f32(
        false, af, false, bpos, (short)0, acc, false, false);
#pragma unroll
    for (int r = 0; r < 8; ++r) {
      int row = 16 * t + hf * 8 + r;
      float v = acc[r] + bp1[row];
      v = v > 0.f ? v : 0.f;
      s_bop[p][row][l15] = (__bf16)v;
    }
  }
  __syncthreads();

  // step 3: pe = wp2 @ pe_h + bp2  (bf16 WMMA, K=64 -> 2 steps)
  float pe_reg[4][8];
#pragma unroll
  for (int mt = 0; mt < 4; ++mt) {
    v8f acc = {};
#pragma unroll
    for (int s = 0; s < 2; ++s) {
      v16bf a, bb;
      int m = 16 * mt + l15;
#pragma unroll
      for (int e = 0; e < 16; ++e) {
        int kk = s * 32 + hf * 16 + e;
        a[e]  = wp2bf[m * Dd + kk];
        bb[e] = s_bop[p][s * 32 + hf * 16 + e][l15];
      }
      acc = __builtin_amdgcn_wmma_f32_16x16x32_bf16(
          false, a, false, bb, (short)0, acc, false, false);
    }
#pragma unroll
    for (int r = 0; r < 8; ++r) pe_reg[mt][r] = acc[r] + bp2[16 * mt + hf * 8 + r];
  }
  __syncthreads();   // done reading pe_h from s_bop

  // write pe (f32) and qk_rel+pe (bf16 B operand)
  const int nb = s_idx[p][l15];
#pragma unroll
  for (int mt = 0; mt < 4; ++mt)
#pragma unroll
    for (int r = 0; r < 8; ++r) {
      int row = 16 * mt + hf * 8 + r;
      float pe = pe_reg[mt][r];
      s_pe[p][row][l15] = pe;
      float qk = qmat[((long)b * Dd + row) * Nn + n]
               - kmat[((long)b * Dd + row) * Nn + nb];
      s_bop[p][row][l15] = (__bf16)(qk + pe);
    }
  __syncthreads();

  // step 4: hidden = relu(wa1 @ (qk+pe) + ba1)  (256x16, 16 M-tiles x 2 K)
#pragma unroll
  for (int mt = 0; mt < 16; ++mt) {
    v8f acc = {};
#pragma unroll
    for (int s = 0; s < 2; ++s) {
      v16bf a, bb;
      int m = 16 * mt + l15;
#pragma unroll
      for (int e = 0; e < 16; ++e) {
        int kk = s * 32 + hf * 16 + e;
        a[e]  = wa1bf[m * Dd + kk];
        bb[e] = s_bop[p][s * 32 + hf * 16 + e][l15];
      }
      acc = __builtin_amdgcn_wmma_f32_16x16x32_bf16(
          false, a, false, bb, (short)0, acc, false, false);
    }
#pragma unroll
    for (int r = 0; r < 8; ++r) {
      int row = 16 * mt + hf * 8 + r;
      float v = acc[r] + ba1[row];
      v = v > 0.f ? v : 0.f;
      s_hid[p][row][l15] = (__bf16)v;
    }
  }
  __syncthreads();

  // step 5: attn = wa2 @ hidden + ba2   (64x16, 4 M-tiles x 8 K)
  float at[4][8];
#pragma unroll
  for (int mt = 0; mt < 4; ++mt) {
    v8f acc = {};
#pragma unroll
    for (int s = 0; s < 8; ++s) {
      v16bf a, bb;
      int m = 16 * mt + l15;
#pragma unroll
      for (int e = 0; e < 16; ++e) {
        int kk = s * 32 + hf * 16 + e;
        a[e]  = wa2bf[m * AHc + kk];
        bb[e] = s_hid[p][s * 32 + hf * 16 + e][l15];
      }
      acc = __builtin_amdgcn_wmma_f32_16x16x32_bf16(
          false, a, false, bb, (short)0, acc, false, false);
    }
#pragma unroll
    for (int r = 0; r < 8; ++r) at[mt][r] = acc[r] + ba2[16 * mt + hf * 8 + r];
  }

  // step 6+7: softmax over the 16 neighbor columns (16-lane half reduction),
  // then agg = sum_k attn * (v + pe)
#pragma unroll
  for (int mt = 0; mt < 4; ++mt)
#pragma unroll
    for (int r = 0; r < 8; ++r) {
      int row = 16 * mt + hf * 8 + r;
      float x = at[mt][r];
      float mx = x;
#pragma unroll
      for (int msk = 1; msk < 16; msk <<= 1) mx = fmaxf(mx, __shfl_xor(mx, msk, 32));
      float e = __expf(x - mx);
      float sm = e;
#pragma unroll
      for (int msk = 1; msk < 16; msk <<= 1) sm += __shfl_xor(sm, msk, 32);
      float a = e / sm;
      float v2 = vmat[((long)b * Dd + row) * Nn + n] + s_pe[p][row][l15];
      float pr = a * v2;
#pragma unroll
      for (int msk = 1; msk < 16; msk <<= 1) pr += __shfl_xor(pr, msk, 32);
      if (l15 == 0) aggbf[((long)b * Dd + row) * Nn + n] = (__bf16)pr;
    }
}

// ---------------- host-side orchestration ----------------
extern "C" void kernel_launch(void* const* d_in, const int* in_sizes, int n_in,
                              void* d_out, int out_size, void* d_ws, size_t ws_size,
                              hipStream_t stream) {
  const float* pos   = (const float*)d_in[0];
  const float* key   = (const float*)d_in[1];
  const float* query = (const float*)d_in[2];
  const float* w1 = (const float*)d_in[3];  const float* b1 = (const float*)d_in[4];
  const float* w2 = (const float*)d_in[5];  const float* b2 = (const float*)d_in[6];
  const float* wsm= (const float*)d_in[7];  const float* bs = (const float*)d_in[8];
  const float* wk = (const float*)d_in[9];  const float* bk = (const float*)d_in[10];
  const float* wq = (const float*)d_in[11]; const float* bq = (const float*)d_in[12];
  const float* wv = (const float*)d_in[13]; const float* bv = (const float*)d_in[14];
  const float* wp1= (const float*)d_in[15]; const float* bp1= (const float*)d_in[16];
  const float* wp2= (const float*)d_in[17]; const float* bp2= (const float*)d_in[18];
  const float* wa1= (const float*)d_in[19]; const float* ba1= (const float*)d_in[20];
  const float* wa2= (const float*)d_in[21]; const float* ba2= (const float*)d_in[22];
  const float* we = (const float*)d_in[23]; const float* be = (const float*)d_in[24];
  float* out = (float*)d_out;

  char* base = (char*)d_ws; size_t off = 0;
  auto bump = [&](size_t bytes) -> void* {
    off = (off + 255) & ~(size_t)255;
    void* p = base + off; off += bytes; return p;
  };
  __bf16* kq_bf   = (__bf16*)bump((size_t)Bc * 2 * Cc * Nn * 2);
  __bf16* h_bf    = (__bf16*)bump((size_t)Bc * Cc * Nn * 2);
  float*  sres    = (float*) bump((size_t)Bc * Cc * Nn * 4);
  float*  valueF  = (float*) bump((size_t)Bc * Cc * Nn * 4);
  __bf16* valueB  = (__bf16*)bump((size_t)Bc * Cc * Nn * 2);
  float*  kmat    = (float*) bump((size_t)Bc * Dd * Nn * 4);
  float*  qmat    = (float*) bump((size_t)Bc * Dd * Nn * 4);
  float*  vmat    = (float*) bump((size_t)Bc * Dd * Nn * 4);
  int*    idx     = (int*)   bump((size_t)Bc * Nn * Kn * 4);
  __bf16* aggbf   = (__bf16*)bump((size_t)Bc * Dd * Nn * 2);
  __bf16* w1bf  = (__bf16*)bump((size_t)Cc * 2 * Cc * 2);
  __bf16* wsbf  = (__bf16*)bump((size_t)Cc * 2 * Cc * 2);
  __bf16* w2bf  = (__bf16*)bump((size_t)Cc * Cc * 2);
  __bf16* wkbf  = (__bf16*)bump((size_t)Dd * Cc * 2);
  __bf16* wqbf  = (__bf16*)bump((size_t)Dd * Cc * 2);
  __bf16* wvbf  = (__bf16*)bump((size_t)Dd * Cc * 2);
  __bf16* wp2bf = (__bf16*)bump((size_t)Dd * PHc * 2);
  __bf16* wa1bf = (__bf16*)bump((size_t)AHc * Dd * 2);
  __bf16* wa2bf = (__bf16*)bump((size_t)Dd * AHc * 2);
  __bf16* webf  = (__bf16*)bump((size_t)Cc * Dd * 2);

  auto cast = [&](const float* s, __bf16* d, long n) {
    castf2bf_kernel<<<dim3((unsigned)((n + 255) / 256)), 256, 0, stream>>>(s, d, n);
  };
  cast(w1, w1bf,  (long)Cc * 2 * Cc);
  cast(wsm, wsbf, (long)Cc * 2 * Cc);
  cast(w2, w2bf,  (long)Cc * Cc);
  cast(wk, wkbf,  (long)Dd * Cc);
  cast(wq, wqbf,  (long)Dd * Cc);
  cast(wv, wvbf,  (long)Dd * Cc);
  cast(wp2, wp2bf,(long)Dd * PHc);
  cast(wa1, wa1bf,(long)AHc * Dd);
  cast(wa2, wa2bf,(long)Dd * AHc);
  cast(we, webf,  (long)Cc * Dd);

  {
    long total = (long)Bc * 2 * Cc * Nn;
    cast_concat_kernel<<<dim3((unsigned)((total + 255) / 256)), 256, 0, stream>>>(
        key, query, kq_bf);
  }

  const long kq_bs  = (long)2 * Cc * Nn;
  const long c_bs   = (long)Cc * Nn;
  const long d_bs   = (long)Dd * Nn;
  dim3 blk(128);

  // h = relu(W1 @ kq + b1)  -> bf16
  gemm_bf16_wmma<<<dim3(Nn / 64, Cc / 64, Bc), blk, 0, stream>>>(
      w1bf, kq_bf, b1, nullptr, nullptr, h_bf, Cc, 2 * Cc, Nn, kq_bs, c_bs, 1);
  // sres = Ws @ kq + bs  -> f32
  gemm_bf16_wmma<<<dim3(Nn / 64, Cc / 64, Bc), blk, 0, stream>>>(
      wsbf, kq_bf, bs, nullptr, sres, nullptr, Cc, 2 * Cc, Nn, kq_bs, c_bs, 0);
  // value = W2 @ h + b2 + sres  -> f32 (identity) + bf16
  gemm_bf16_wmma<<<dim3(Nn / 64, Cc / 64, Bc), blk, 0, stream>>>(
      w2bf, h_bf, b2, sres, valueF, valueB, Cc, Cc, Nn, c_bs, c_bs, 0);
  // k/q/v projections (D=64 -> one M tile row)
  gemm_bf16_wmma<<<dim3(Nn / 64, Dd / 64, Bc), blk, 0, stream>>>(
      wkbf, kq_bf, bk, nullptr, kmat, nullptr, Dd, Cc, Nn, kq_bs, d_bs, 0);
  gemm_bf16_wmma<<<dim3(Nn / 64, Dd / 64, Bc), blk, 0, stream>>>(
      wqbf, kq_bf + (long)Cc * Nn, bq, nullptr, qmat, nullptr, Dd, Cc, Nn, kq_bs, d_bs, 0);
  gemm_bf16_wmma<<<dim3(Nn / 64, Dd / 64, Bc), blk, 0, stream>>>(
      wvbf, valueB, bv, nullptr, vmat, nullptr, Dd, Cc, Nn, c_bs, d_bs, 0);

  // KNN top-16
  knn_kernel<<<dim3(Nn / 256, Bc), 256, 0, stream>>>(pos, idx);

  // fused per-point attention -> agg bf16
  attn_point_kernel<<<dim3(Nn / 4, Bc), 128, 0, stream>>>(
      pos, idx, qmat, kmat, vmat, wp1, bp1, wp2bf, bp2,
      wa1bf, ba1, wa2bf, ba2, aggbf);

  // out = we @ agg + be + identity(value)
  gemm_bf16_wmma<<<dim3(Nn / 64, Cc / 64, Bc), blk, 0, stream>>>(
      webf, aggbf, be, valueF, out, nullptr, Cc, Dd, Nn, d_bs, c_bs, 0);
}
